// GPT2_52158082843259
// MI455X (gfx1250) — matrix operand order repaired and checked by reading, not compiled, not run on hardware
//
#include <hip/hip_runtime.h>
#include <hip/hip_bf16.h>

// ---------------------------------------------------------------------------
// GPT-2 small forward for MI455X (gfx1250, wave32).
// Memory-bound workload (~0.9 GB HBM traffic, ~580 GFLOP): bf16 WMMA
// (v_wmma_f32_16x16x32_bf16) with f32 accumulation; fp32 weights converted to
// bf16 while staging into LDS; 192MB L2 gives weight-tile reuse across the 16
// M-blocks. A-operand staging uses gfx1250 async global->LDS copies when the
// toolchain exposes the builtins (ASYNCcnt path), else a vectorized fallback.
// ---------------------------------------------------------------------------

typedef unsigned short u16;
typedef __attribute__((ext_vector_type(16))) __bf16 v16bf;
typedef __attribute__((ext_vector_type(8)))  float  v8f;

#if defined(__gfx1250__) && __has_builtin(__builtin_amdgcn_global_load_async_to_lds_b128) && __has_builtin(__builtin_amdgcn_s_wait_asynccnt)
#define HAVE_ASYNC_LDS 1
// exact parameter types per hipcc diagnostic: pointer to int vector_size(16)
typedef int v4i_t __attribute__((vector_size(4 * sizeof(int))));
typedef __attribute__((address_space(1))) v4i_t* gas1_b128;
typedef __attribute__((address_space(3))) v4i_t* las3_b128;
#else
#define HAVE_ASYNC_LDS 0
#endif

#define GPT_V  50257
#define GPT_P  1024
#define GPT_E  768
#define GPT_L  12
#define GPT_H  12
#define GPT_HD 64
#define GPT_FF 3072
#define GPT_B  2
#define GPT_S  1024
#define GPT_M  (GPT_B * GPT_S)   // 2048 rows

struct U8 { unsigned int v[8]; };

// float -> bf16, round-to-nearest-even
__device__ __forceinline__ u16 f2bf(float f) {
    unsigned int u = __builtin_bit_cast(unsigned int, f);
    u += 0x7FFFu + ((u >> 16) & 1u);
    return (u16)(u >> 16);
}

// pack two bf16 (lo = even k, hi = odd k) into one dword for 32-bit DS stores
__device__ __forceinline__ unsigned int packbf(float lo, float hi) {
    return (unsigned int)f2bf(lo) | ((unsigned int)f2bf(hi) << 16);
}

// A-fragment (16x32 bf16) from row-major LDS tile; `lds` points at (row0,k0).
// Layout (ISA 7.12.2): lane m = lane%16; half = lane/16;
// VGPR j<4 holds k = 2j + 8*half ; VGPR j>=4 holds k = 16 + 2(j-4) + 8*half.
__device__ __forceinline__ v16bf lds_load_a16(const u16* lds, int ld) {
    int lane = threadIdx.x & 31;
    const u16* p = lds + (lane & 15) * ld + ((lane >> 4) << 3);
    U8 t;
#pragma unroll
    for (int j = 0; j < 4; ++j) {
        t.v[j]     = *reinterpret_cast<const unsigned int*>(p + 2 * j);
        t.v[j + 4] = *reinterpret_cast<const unsigned int*>(p + 16 + 2 * j);
    }
    return __builtin_bit_cast(v16bf, t);
}

// B-fragment (32x16 bf16) from n-major LDS storage [n][k]; `lds` at (n0,k0).
// Layout: lane n = lane%16; half = lane/16; VGPR j<4 holds k = 2j + 8*half ; VGPR j>=4 holds k = 16 + 2(j-4) + 8*half.
__device__ __forceinline__ v16bf lds_load_b16(const u16* lds, int ld) {
    int lane = threadIdx.x & 31;
    const u16* p = lds + (lane & 15) * ld + ((lane >> 4) << 3);
    U8 t;
#pragma unroll
    for (int j = 0; j < 4; ++j) {
        t.v[j]     = *reinterpret_cast<const unsigned int*>(p + 2 * j);
        t.v[j + 4] = *reinterpret_cast<const unsigned int*>(p + 16 + 2 * j);
    }
    return __builtin_bit_cast(v16bf, t);
}

__device__ __forceinline__ v8f wmma_bf16(v16bf a, v16bf b, v8f c) {
    return __builtin_amdgcn_wmma_f32_16x16x32_bf16(
        /*neg_a=*/false, a, /*neg_b=*/false, b,
        /*c_mod=*/(short)0, c, /*reuse_a=*/false, /*reuse_b=*/false);
}

__device__ __forceinline__ float gelu_tanh(float u) {
    const float c = 0.7978845608028654f;  // sqrt(2/pi)
    float t = tanhf(c * (u + 0.044715f * u * u * u));
    return 0.5f * u * (1.0f + t);
}

// ---------------------------------------------------------------------------
// Embedding: h[m,e] = tok_embed[x[m],e] + pos_embed[m%S,e]
// ---------------------------------------------------------------------------
__global__ __launch_bounds__(256) void embed_kernel(
    const int* __restrict__ x, const float* __restrict__ tok,
    const float* __restrict__ pos, float* __restrict__ h) {
    int row = blockIdx.x;
    int s = row & (GPT_S - 1);
    int t = x[row];
    const float* tr = tok + (size_t)t * GPT_E;
    const float* pr = pos + (size_t)s * GPT_E;
    float* hr = h + (size_t)row * GPT_E;
#pragma unroll
    for (int i = 0; i < 3; ++i) {
        int e = threadIdx.x + i * 256;
        hr[e] = tr[e] + pr[e];
    }
}

// ---------------------------------------------------------------------------
// LayerNorm over E=768, one block per row, output bf16 (GEMM A-operand).
// ---------------------------------------------------------------------------
__global__ __launch_bounds__(256) void ln_bf16_kernel(
    const float* __restrict__ h, const float* __restrict__ gs,
    const float* __restrict__ gb, u16* __restrict__ out) {
    int row = blockIdx.x, tid = threadIdx.x;
    const float* hr = h + (size_t)row * GPT_E;
    float x[3], s = 0.f, q = 0.f;
#pragma unroll
    for (int i = 0; i < 3; ++i) {
        x[i] = hr[tid + i * 256];
        s += x[i];
        q += x[i] * x[i];
    }
#pragma unroll
    for (int off = 16; off; off >>= 1) {
        s += __shfl_xor(s, off);
        q += __shfl_xor(q, off);
    }
    __shared__ float rs[8], rq[8];
    int w = tid >> 5, lane = tid & 31;
    if (lane == 0) { rs[w] = s; rq[w] = q; }
    __syncthreads();
    float S = 0.f, Q = 0.f;
#pragma unroll
    for (int j = 0; j < 8; ++j) { S += rs[j]; Q += rq[j]; }
    float mean = S * (1.0f / GPT_E);
    float var  = Q * (1.0f / GPT_E) - mean * mean;
    float rstd = rsqrtf(var + 1e-5f);
    u16* orow = out + (size_t)row * GPT_E;
#pragma unroll
    for (int i = 0; i < 3; ++i) {
        int e = tid + i * 256;
        orow[e] = f2bf((x[i] - mean) * rstd * gs[e] + gb[e]);
    }
}

// ---------------------------------------------------------------------------
// Tiled WMMA GEMM: C[M,N] = A_bf16[M,K] * B_f32 (+bias, + optional fusion).
//   BT=false: B is [K,N] row-major (all layer weights).
//   BT=true : B is [N,K] row-major (tok_embed for the LM head).
// Block tile 128x128x32, 256 threads = 8 waves (4 m x 2 n), wave = 32x64.
// EPI: 0 = f32 out = acc+bias          (qkv)
//      1 = f32 out = resid+acc+bias    (out-proj / fc2, residual fused)
//      2 = bf16 out = gelu(acc+bias)   (fc1)
//      3 = f32 out = acc               (LM head, N-guarded)
// ---------------------------------------------------------------------------
template <int EPI, bool BT>
__global__ __launch_bounds__(256) void gemm_wmma_kernel(
    const u16* __restrict__ A, const float* __restrict__ Bg,
    const float* __restrict__ bias, void* __restrict__ Cout,
    const float* __restrict__ resid, int Mdim, int Ndim, int Kdim) {
    __shared__ u16 As[128 * 32];   // [m][k] row-major
    __shared__ u16 Bs[128 * 32];   // [n][k] n-major (aligned 32b k-pair reads)

    int tid = threadIdx.x;
    int n0 = blockIdx.x * 128;
    int m0 = blockIdx.y * 128;
    int w = tid >> 5;
    int wm = w >> 1;     // 0..3
    int wn = w & 1;      // 0..1
    int lane = tid & 31;

    v8f acc[2][4] = {};

    for (int k0 = 0; k0 < Kdim; k0 += 32) {
        __syncthreads();
        // ---- stage A tile (bf16, 128x32): async global->LDS when available
#if HAVE_ASYNC_LDS
#pragma unroll
        for (int i = 0; i < 2; ++i) {
            int idx = tid + i * 256;
            int row = idx >> 2, c4 = idx & 3;
            const u16* gsrc = A + (size_t)(m0 + row) * Kdim + k0 + c4 * 8;
            u16* ldst = As + idx * 8;
            __builtin_amdgcn_global_load_async_to_lds_b128(
                (gas1_b128)(void*)gsrc, (las3_b128)(void*)ldst, 0, 0);
        }
#else
#pragma unroll
        for (int i = 0; i < 2; ++i) {
            int idx = tid + i * 256;
            int row = idx >> 2, c4 = idx & 3;
            uint4 val = *reinterpret_cast<const uint4*>(
                A + (size_t)(m0 + row) * Kdim + k0 + c4 * 8);
            reinterpret_cast<uint4*>(As)[idx] = val;
        }
#endif
        // ---- stage B tile (f32 -> bf16), write n-major, packed k-pairs ----
        if constexpr (!BT) {
#pragma unroll
            for (int i = 0; i < 2; ++i) {
                int idx = tid + i * 256;          // 0..511
                int kp = idx >> 5, c = idx & 31;  // k-pair row, float4 col
                const float4 v0 = *reinterpret_cast<const float4*>(
                    Bg + (size_t)(k0 + 2 * kp) * Ndim + n0 + c * 4);
                const float4 v1 = *reinterpret_cast<const float4*>(
                    Bg + (size_t)(k0 + 2 * kp + 1) * Ndim + n0 + c * 4);
                int nb = c * 4;
                *reinterpret_cast<unsigned int*>(Bs + (nb + 0) * 32 + 2 * kp) = packbf(v0.x, v1.x);
                *reinterpret_cast<unsigned int*>(Bs + (nb + 1) * 32 + 2 * kp) = packbf(v0.y, v1.y);
                *reinterpret_cast<unsigned int*>(Bs + (nb + 2) * 32 + 2 * kp) = packbf(v0.z, v1.z);
                *reinterpret_cast<unsigned int*>(Bs + (nb + 3) * 32 + 2 * kp) = packbf(v0.w, v1.w);
            }
        } else {
#pragma unroll
            for (int i = 0; i < 4; ++i) {
                int idx = tid + i * 256;
                int row = idx >> 3, c = idx & 7;  // n-row, float4 within k
                int n = n0 + row;
                float4 vv = make_float4(0.f, 0.f, 0.f, 0.f);
                if (n < Ndim)
                    vv = *reinterpret_cast<const float4*>(
                        Bg + (size_t)n * Kdim + k0 + c * 4);
                *reinterpret_cast<uint2*>(Bs + row * 32 + c * 4) =
                    make_uint2(packbf(vv.x, vv.y), packbf(vv.z, vv.w));
            }
        }
#if HAVE_ASYNC_LDS
        __builtin_amdgcn_s_wait_asynccnt(0);
#endif
        __syncthreads();

        // prefetch next weight tile (gfx1250 global_prefetch_b8 hint)
        if (k0 + 32 < Kdim) {
            size_t nb = BT ? ((size_t)n0 * Kdim + k0 + 32)
                           : ((size_t)(k0 + 32) * Ndim + n0);
            __builtin_prefetch(Bg + nb + tid * 4);
        }

        // ---- compute: 8 WMMAs per wave per K-step ----
        v16bf af0 = lds_load_a16(As + (wm * 32 +  0) * 32, 32);
        v16bf af1 = lds_load_a16(As + (wm * 32 + 16) * 32, 32);
#pragma unroll
        for (int fn = 0; fn < 4; ++fn) {
            v16bf bf = lds_load_b16(Bs + (wn * 64 + fn * 16) * 32, 32);
            acc[0][fn] = wmma_bf16(af0, bf, acc[0][fn]);
            acc[1][fn] = wmma_bf16(af1, bf, acc[1][fn]);
        }
    }

    // ---- epilogue: C layout lane->(n=lane%16, m=r+8*(lane/16)) ----
    int cn = lane & 15;
    int mo = (lane >> 4) * 8;
#pragma unroll
    for (int fm = 0; fm < 2; ++fm) {
#pragma unroll
        for (int fn = 0; fn < 4; ++fn) {
            int nc = n0 + wn * 64 + fn * 16 + cn;
            if (nc >= Ndim) continue;
            float bv = 0.f;
            if constexpr (EPI != 3) bv = bias[nc];
#pragma unroll
            for (int r = 0; r < 8; ++r) {
                int mr = m0 + wm * 32 + fm * 16 + mo + r;
                float v = acc[fm][fn][r] + bv;
                size_t ci = (size_t)mr * Ndim + nc;
                if constexpr (EPI == 0) {
                    reinterpret_cast<float*>(Cout)[ci] = v;
                } else if constexpr (EPI == 1) {
                    reinterpret_cast<float*>(Cout)[ci] = resid[ci] + v;
                } else if constexpr (EPI == 2) {
                    reinterpret_cast<u16*>(Cout)[ci] = f2bf(gelu_tanh(v));
                } else {
                    reinterpret_cast<float*>(Cout)[ci] = v;
                }
            }
        }
    }
}

// ---------------------------------------------------------------------------
// Flash-style causal attention with WMMA. One block = (b,h, 64-query tile),
// 128 threads = 4 waves; each wave owns 16 query rows. K/V streamed in
// 64-key tiles; online softmax; P round-trips through per-wave LDS so its
// C-layout can be reloaded in A-layout for P*V.
// ---------------------------------------------------------------------------
__global__ __launch_bounds__(128) void attn_wmma_kernel(
    const float* __restrict__ qkv, u16* __restrict__ z) {
    __shared__ u16 Qs[64 * 64];       // [q][d]
    __shared__ u16 Ks[64 * 64];       // [key][d]  (n-major for S=Q*K^T)
    __shared__ u16 Vt[64 * 64];       // [d][key]  (n-major for O=P*V)
    __shared__ u16 Ps[4 * 16 * 64];   // per-wave P tile [16][64]

    const int qb = blockIdx.x;            // 0..15
    const int bh = blockIdx.y;            // 0..23
    const int b = bh / GPT_H, hh = bh % GPT_H;
    const int tid = threadIdx.x;
    const int lane = tid & 31;
    const int w = tid >> 5;               // 0..3
    const int E3 = 3 * GPT_E;

    // ---- stage Q tile (f32 -> bf16) ----
#pragma unroll
    for (int i = 0; i < 8; ++i) {
        int idx = tid + i * 128;          // 0..1023 float4s
        int row = idx >> 4, c = idx & 15;
        int s = qb * 64 + row;
        float4 vv = *reinterpret_cast<const float4*>(
            qkv + (size_t)(b * GPT_S + s) * E3 + hh * GPT_HD + c * 4);
        *reinterpret_cast<uint2*>(Qs + row * 64 + c * 4) =
            make_uint2(packbf(vv.x, vv.y), packbf(vv.z, vv.w));
    }
    __syncthreads();

    v16bf Aq[2];
    Aq[0] = lds_load_a16(Qs + (w * 16) * 64 +  0, 64);
    Aq[1] = lds_load_a16(Qs + (w * 16) * 64 + 32, 64);

    float mrun[8], lrun[8];
#pragma unroll
    for (int r = 0; r < 8; ++r) { mrun[r] = -3.0e38f; lrun[r] = 0.f; }
    v8f Of[4] = {};

    for (int kb = 0; kb <= qb; ++kb) {
        __syncthreads();
        // ---- stage K (row-major, packed along d) ----
#pragma unroll
        for (int i = 0; i < 8; ++i) {
            int idx = tid + i * 128;      // 0..1023
            int row = idx >> 4, c = idx & 15;
            int s = kb * 64 + row;
            float4 vv = *reinterpret_cast<const float4*>(
                qkv + (size_t)(b * GPT_S + s) * E3 + GPT_E + hh * GPT_HD + c * 4);
            *reinterpret_cast<uint2*>(Ks + row * 64 + c * 4) =
                make_uint2(packbf(vv.x, vv.y), packbf(vv.z, vv.w));
        }
        // ---- stage V transposed [d][key], packed key-pairs ----
#pragma unroll
        for (int i = 0; i < 4; ++i) {
            int idx = tid + i * 128;      // 0..511
            int kp = idx >> 4, c = idx & 15;   // key pair, float4 of dims
            int s = kb * 64 + 2 * kp;
            const float* vp = qkv + (size_t)(b * GPT_S + s) * E3 + 2 * GPT_E + hh * GPT_HD + c * 4;
            float4 a = *reinterpret_cast<const float4*>(vp);
            float4 bb = *reinterpret_cast<const float4*>(vp + E3);
            *reinterpret_cast<unsigned int*>(Vt + (c * 4 + 0) * 64 + 2 * kp) = packbf(a.x, bb.x);
            *reinterpret_cast<unsigned int*>(Vt + (c * 4 + 1) * 64 + 2 * kp) = packbf(a.y, bb.y);
            *reinterpret_cast<unsigned int*>(Vt + (c * 4 + 2) * 64 + 2 * kp) = packbf(a.z, bb.z);
            *reinterpret_cast<unsigned int*>(Vt + (c * 4 + 3) * 64 + 2 * kp) = packbf(a.w, bb.w);
        }
        __syncthreads();

        // ---- S = Q * K^T / sqrt(HD), causal mask on diagonal tile ----
        v8f sf[4];
#pragma unroll
        for (int fn = 0; fn < 4; ++fn) {
            v8f s = {};
#pragma unroll
            for (int ks = 0; ks < 2; ++ks) {
                v16bf Bk = lds_load_b16(Ks + (fn * 16) * 64 + ks * 32, 64);
                s = wmma_bf16(Aq[ks], Bk, s);
            }
            s = s * 0.125f;               // 1/sqrt(64)
            if (kb == qb) {
                int jcol = kb * 64 + fn * 16 + (lane & 15);
#pragma unroll
                for (int r = 0; r < 8; ++r) {
                    int irow = qb * 64 + w * 16 + (lane >> 4) * 8 + r;
                    if (jcol > irow) s[r] = -10000.0f;
                }
            }
            sf[fn] = s;
        }

        // ---- online softmax per query row (rows split across lane halves) --
#pragma unroll
        for (int r = 0; r < 8; ++r) {
            float mloc = fmaxf(fmaxf(sf[0][r], sf[1][r]),
                               fmaxf(sf[2][r], sf[3][r]));
#pragma unroll
            for (int off = 8; off; off >>= 1)
                mloc = fmaxf(mloc, __shfl_xor(mloc, off, 16));
            float mnew = fmaxf(mrun[r], mloc);
            float corr = __expf(mrun[r] - mnew);
            mrun[r] = mnew;
            float psum = 0.f;
#pragma unroll
            for (int fn = 0; fn < 4; ++fn) {
                float p = __expf(sf[fn][r] - mnew);
                sf[fn][r] = p;
                psum += p;
            }
#pragma unroll
            for (int off = 8; off; off >>= 1)
                psum += __shfl_xor(psum, off, 16);
            lrun[r] = lrun[r] * corr + psum;
#pragma unroll
            for (int fn = 0; fn < 4; ++fn) Of[fn][r] *= corr;
        }

        // ---- P (bf16) -> per-wave LDS; same-wave DS ops stay in order ----
        u16* Pw = Ps + w * 16 * 64;
#pragma unroll
        for (int fn = 0; fn < 4; ++fn)
#pragma unroll
            for (int r = 0; r < 8; ++r)
                Pw[((lane >> 4) * 8 + r) * 64 + fn * 16 + (lane & 15)] =
                    f2bf(sf[fn][r]);

        // ---- O += P * V ----
#pragma unroll
        for (int ks = 0; ks < 2; ++ks) {
            v16bf Ap = lds_load_a16(Pw + ks * 32, 64);
#pragma unroll
            for (int fn = 0; fn < 4; ++fn) {
                v16bf Bv = lds_load_b16(Vt + (fn * 16) * 64 + ks * 32, 64);
                Of[fn] = wmma_bf16(Ap, Bv, Of[fn]);
            }
        }
    }

    // ---- normalize and write z (bf16) ----
    float inv[8];
#pragma unroll
    for (int r = 0; r < 8; ++r) inv[r] = 1.0f / lrun[r];
#pragma unroll
    for (int fn = 0; fn < 4; ++fn)
#pragma unroll
        for (int r = 0; r < 8; ++r) {
            int srow = qb * 64 + w * 16 + (lane >> 4) * 8 + r;
            int col  = hh * GPT_HD + fn * 16 + (lane & 15);
            z[(size_t)(b * GPT_S + srow) * GPT_E + col] = f2bf(Of[fn][r] * inv[r]);
        }
}

// ---------------------------------------------------------------------------
// Host side
// ---------------------------------------------------------------------------
extern "C" void kernel_launch(void* const* d_in, const int* in_sizes, int n_in,
                              void* d_out, int out_size, void* d_ws, size_t ws_size,
                              hipStream_t stream) {
    (void)in_sizes; (void)n_in; (void)out_size; (void)ws_size;

    const int*   x     = (const int*)  d_in[0];
    const float* tok   = (const float*)d_in[1];
    const float* pos   = (const float*)d_in[2];
    const float* ln1_s = (const float*)d_in[3];
    const float* ln1_b = (const float*)d_in[4];
    const float* qkv_w = (const float*)d_in[5];
    const float* qkv_b = (const float*)d_in[6];
    const float* out_w = (const float*)d_in[7];
    const float* out_b = (const float*)d_in[8];
    const float* ln2_s = (const float*)d_in[9];
    const float* ln2_b = (const float*)d_in[10];
    const float* fc1_w = (const float*)d_in[11];
    const float* fc1_b = (const float*)d_in[12];
    const float* fc2_w = (const float*)d_in[13];
    const float* fc2_b = (const float*)d_in[14];
    const float* lnf_s = (const float*)d_in[15];
    const float* lnf_b = (const float*)d_in[16];

    // workspace layout
    char* wsb = (char*)d_ws;
    constexpr size_t hBytes    = (size_t)GPT_M * GPT_E  * 4;   //  6.29 MB f32
    constexpr size_t qkvBytes  = (size_t)GPT_M * 3*GPT_E* 4;   // 18.87 MB f32
    constexpr size_t xnBytes   = (size_t)GPT_M * GPT_E  * 2;   //  3.15 MB bf16
    constexpr size_t zBytes    = (size_t)GPT_M * GPT_E  * 2;   //  3.15 MB bf16
    float* h    = (float*)(wsb);
    float* qkvb = (float*)(wsb + hBytes);
    u16*   xn   = (u16*)  (wsb + hBytes + qkvBytes);
    u16*   zb   = (u16*)  (wsb + hBytes + qkvBytes + xnBytes);
    u16*   ffa  = (u16*)  (wsb + hBytes + qkvBytes + xnBytes + zBytes);

    embed_kernel<<<GPT_M, 256, 0, stream>>>(x, tok, pos, h);

    for (int l = 0; l < GPT_L; ++l) {
        const float* qw = qkv_w + (size_t)l * GPT_E * 3 * GPT_E;
        const float* qB = qkv_b + (size_t)l * 3 * GPT_E;
        const float* ow = out_w + (size_t)l * GPT_E * GPT_E;
        const float* oB = out_b + (size_t)l * GPT_E;
        const float* w1 = fc1_w + (size_t)l * GPT_E * GPT_FF;
        const float* b1 = fc1_b + (size_t)l * GPT_FF;
        const float* w2 = fc2_w + (size_t)l * GPT_FF * GPT_E;
        const float* b2 = fc2_b + (size_t)l * GPT_E;

        ln_bf16_kernel<<<GPT_M, 256, 0, stream>>>(
            h, ln1_s + l * GPT_E, ln1_b + l * GPT_E, xn);

        gemm_wmma_kernel<0, false><<<dim3(3 * GPT_E / 128, GPT_M / 128),
                                     256, 0, stream>>>(
            xn, qw, qB, qkvb, nullptr, GPT_M, 3 * GPT_E, GPT_E);

        attn_wmma_kernel<<<dim3(GPT_S / 64, GPT_B * GPT_H), 128, 0, stream>>>(
            qkvb, zb);

        gemm_wmma_kernel<1, false><<<dim3(GPT_E / 128, GPT_M / 128),
                                     256, 0, stream>>>(
            zb, ow, oB, h, h, GPT_M, GPT_E, GPT_E);

        ln_bf16_kernel<<<GPT_M, 256, 0, stream>>>(
            h, ln2_s + l * GPT_E, ln2_b + l * GPT_E, xn);

        gemm_wmma_kernel<2, false><<<dim3(GPT_FF / 128, GPT_M / 128),
                                     256, 0, stream>>>(
            xn, w1, b1, ffa, nullptr, GPT_M, GPT_FF, GPT_E);

        gemm_wmma_kernel<1, false><<<dim3(GPT_E / 128, GPT_M / 128),
                                     256, 0, stream>>>(
            ffa, w2, b2, h, h, GPT_M, GPT_E, GPT_FF);
    }

    ln_bf16_kernel<<<GPT_M, 256, 0, stream>>>(h, lnf_s, lnf_b, xn);

    // LM head: logits[m,v] = sum_e xn[m,e] * tok_embed[v,e]  (B transposed)
    gemm_wmma_kernel<3, true><<<dim3((GPT_V + 127) / 128, GPT_M / 128),
                                256, 0, stream>>>(
        xn, tok, nullptr, (float*)d_out, nullptr, GPT_M, GPT_V, GPT_E);
}
